// GraphSAGE_42752104464586
// MI455X (gfx1250) — matrix operand order repaired and checked by reading, not compiled, not run on hardware
//
#include <hip/hip_runtime.h>
#include <hip/hip_bf16.h>

// GraphSAGE (3-layer, mean aggr) on MI455X / gfx1250.
// adj (1 GB) read once (non-temporal) -> 32 MB bitmask (L2-resident).
// SpMM via bf16 WMMA: one bit-expanded A fragment shared by 8 WMMAs,
// B fragments preloaded for load/WMMA overlap. Dense layers stage weights
// into LDS with global_load_async_to_lds_b128 (ASYNCcnt path).

#define GN   16384          // nodes
#define GNW  (GN / 32)      // 512 bitmask words per target row
#define GF   128            // hidden feature width
#define KWPW (GNW / 8)      // 64 bitmask words per wave (K-split across 8 waves)

typedef __attribute__((ext_vector_type(16))) __bf16 v16bf;
typedef __attribute__((ext_vector_type(8)))  __bf16 v8bf;
typedef __attribute__((ext_vector_type(8)))  float  v8f;

__device__ __forceinline__ unsigned lds_addr_of(void* p) {
  // generic -> LDS address space; numeric value = LDS byte offset
  return (unsigned)(unsigned long long)(__attribute__((address_space(3))) char*)(char*)p;
}

// ---------------------------------------------------------------- utilities
__global__ void __launch_bounds__(256) k_zero_u32(unsigned* __restrict__ p, int n) {
  int i = blockIdx.x * 256 + threadIdx.x;
  if (i < n) p[i] = 0u;
}

// Pack adj (row=source s, col=target t) into target-major bitmask + degree.
__global__ void __launch_bounds__(256) k_pack_mask(const float* __restrict__ adj,
                                                   unsigned* __restrict__ bits,
                                                   unsigned* __restrict__ deg) {
  const int t  = blockIdx.x * 256 + threadIdx.x;  // target column
  const int s0 = blockIdx.y * 32;                 // source row chunk
  unsigned w = 0u;
  #pragma unroll 8
  for (int i = 0; i < 32; ++i) {
    float v = __builtin_nontemporal_load(&adj[(size_t)(s0 + i) * GN + t]); // stream, don't pollute L2
    w |= (v != 0.0f ? 1u : 0u) << i;
  }
  bits[(size_t)t * GNW + (s0 >> 5)] = w;
  if (w) atomicAdd(&deg[t], (unsigned)__popc(w));
}

__global__ void __launch_bounds__(256) k_deg_inv(const unsigned* __restrict__ deg,
                                                 float* __restrict__ dinv) {
  int t = blockIdx.x * 256 + threadIdx.x;
  unsigned d = deg[t];
  dinv[t] = d ? 1.0f / (float)d : 0.0f;
}

// f32 x -> bf16 h (row-major) + hT (feature-major, B-operand layout for SpMM)
__global__ void __launch_bounds__(256) k_cvt_x(const float* __restrict__ x,
                                               __bf16* __restrict__ h,
                                               __bf16* __restrict__ hT) {
  int idx = blockIdx.x * 256 + threadIdx.x;   // over GN*GF
  int t = idx >> 7, f = idx & 127;
  __bf16 b = (__bf16)x[idx];
  h[idx] = b;
  hT[(size_t)f * GN + t] = b;
}

// Wcat[o][k] = k<128 ? Wl[o][k] : Wr[o][k-128]  (bf16)
__global__ void __launch_bounds__(256) k_build_wcat(const float* __restrict__ Wl,
                                                    const float* __restrict__ Wr,
                                                    __bf16* __restrict__ Wcat, int Fo) {
  int idx = blockIdx.x * 256 + threadIdx.x;
  if (idx >= Fo * 256) return;
  int o = idx >> 8, k = idx & 255;
  float v = (k < GF) ? Wl[o * GF + k] : Wr[o * GF + (k - GF)];
  Wcat[idx] = (__bf16)v;
}

// ------------------------------------------------------------------- SpMM
// agg[t][f] = deg_inv[t] * sum_s mask_t[t][s] * h[s][f]
// Block: 16 targets; 8 waves split K (2048 sources each); each wave computes
// all 8 feature tiles reusing one bit-expanded A fragment per K-chunk.
// Cross-wave reduction through 64 KB LDS (deterministic, no atomics).
__global__ void __launch_bounds__(256) k_spmm(const unsigned* __restrict__ bits,
                                              const __bf16* __restrict__ hT,
                                              const float* __restrict__ dinv,
                                              __bf16* __restrict__ agg) {
  __shared__ float red[8][16][128];   // [wave][target row][feature] = 64 KB

  const int lane = threadIdx.x & 31;
  const int wave = threadIdx.x >> 5;
  const int t0 = blockIdx.x << 4;
  const int m = lane & 15;
  const int g = lane >> 4;

  const int kw0 = wave * KWPW;        // this wave's bitmask-word range
  const unsigned* __restrict__ wrow = bits + (size_t)(t0 + m) * GNW + kw0;

  v8f acc[8];
  #pragma unroll
  for (int ft = 0; ft < 8; ++ft) acc[ft] = (v8f){0.f,0.f,0.f,0.f,0.f,0.f,0.f,0.f};

  for (int kc = 0; kc < KWPW; ++kc) {
    unsigned w = wrow[kc];
    __builtin_prefetch(wrow + kc + 16, 0, 0);        // global_prefetch_b8
    // wave-uniform skip of all-zero 16x32 tiles (~60% at p=0.001) -> skips 8 WMMAs
    if (__ballot(w != 0u) == 0ull) continue;

    // Preload all 8 B fragments first so the 16 loads clause up and overlap
    // with A-expansion VALU and WMMA issue (avoids s_wait_loadcnt 0 per WMMA).
    const size_t sbase = ((size_t)(kw0 + kc) << 5) + (g << 4);
    v16bf bfr[8];
    #pragma unroll
    for (int ft = 0; ft < 8; ++ft)
      bfr[ft] = *reinterpret_cast<const v16bf*>(hT + (size_t)((ft << 4) + m) * GN + sbase);

    // A fragment: element e -> K = e + 16g. Build packed bf16 pairs
    // branch-free: 2 bits -> {0,1.0bf16}x2 via ((t*0x8001)&0x10001)*0x3F80.
    union { v16bf v; unsigned u[8]; } au;
    #pragma unroll
    for (int d = 0; d < 8; ++d) {
      int e  = d << 1;
      int k0 = e + (g << 4);
      unsigned t2 = (w >> k0) & 3u;
      au.u[d] = ((t2 * 0x8001u) & 0x10001u) * 0x3F80u;
    }

    #pragma unroll
    for (int ft = 0; ft < 8; ++ft)
      acc[ft] = __builtin_amdgcn_wmma_f32_16x16x32_bf16(false, au.v, false, bfr[ft],
                                                        (short)0, acc[ft], false, false);
  }

  // dump per-wave C fragments to LDS (bank-swizzled on the g half)
  #pragma unroll
  for (int ft = 0; ft < 8; ++ft) {
    int col = ((ft << 4) + m) ^ (g << 3);
    #pragma unroll
    for (int r = 0; r < 8; ++r) red[wave][r + (g << 3)][col] = acc[ft][r];
  }
  __syncthreads();

  // reduce 8 wave-partials; each thread owns one contiguous 8-feature run
  const int tl  = (threadIdx.x * 8) >> 7;       // target row 0..15
  const int fb  = (threadIdx.x * 8) & 127;      // feature base
  const int sw  = (tl & 8) ? 8 : 0;             // undo store swizzle
  const int t   = t0 + tl;
  const float sc = dinv[t];
  v8bf outv;
  #pragma unroll
  for (int p = 0; p < 8; ++p) {
    int col = (fb + p) ^ sw;
    float s = 0.f;
    #pragma unroll
    for (int wv = 0; wv < 8; ++wv) s += red[wv][tl][col];
    outv[p] = (__bf16)(s * sc);
  }
  *reinterpret_cast<v8bf*>(agg + (size_t)t * GF + fb) = outv;
}

// ------------------------------------------------------------------- dense
// out = act( [agg|h] @ Wcat^T + b ) ; K = 256 (8 chunks of 32).
// Wcat staged into LDS via async DMA (global_load_async_to_lds_b128),
// XOR-swizzled at 32 B granularity to spread banks across the m lanes.
template <bool RELU, bool TO_BF16, int FO>
__global__ void __launch_bounds__(256) k_dense(const __bf16* __restrict__ agg,
                                               const __bf16* __restrict__ h,
                                               const __bf16* __restrict__ Wcat,
                                               const float* __restrict__ bias,
                                               __bf16* __restrict__ hout,
                                               __bf16* __restrict__ hToutT,
                                               float* __restrict__ fout) {
  __shared__ __bf16 wsm[FO * 256];    // 64 KB (FO=128) / 32 KB (FO=64)

  const int lane = threadIdx.x & 31;
  const int wave = threadIdx.x >> 5;
  const int t0 = blockIdx.x << 4;
  const int o0 = wave << 4;
  const int m = lane & 15;
  const int g = lane >> 4;

  { // async-stage Wcat (FO*512 bytes) into LDS; 16 B chunks, pair-swizzled
    const unsigned base = lds_addr_of(wsm);
    const char* gsrc = (const char*)Wcat;
    const int nthr = (int)blockDim.x;
    #pragma unroll
    for (int it = 0; it < 16; ++it) {
      int idx = it * nthr + (int)threadIdx.x;           // 16 B chunk id, FO*32 total
      int o  = idx >> 5;                                // weight row
      int c  = idx & 31;                                // 16 B chunk within row
      int p2 = c >> 1, hf = c & 1;                      // 32 B pair + half
      unsigned ldsa = base + (unsigned)(o * 512 + ((((p2 ^ (o & 15)) << 1) | hf) << 4));
      unsigned long long ga = (unsigned long long)(size_t)(gsrc + ((size_t)idx << 4));
      asm volatile("global_load_async_to_lds_b128 %0, %1, off"
                   :: "v"(ldsa), "v"(ga) : "memory");
    }
    asm volatile("s_wait_asynccnt 0x0" ::: "memory");
    __syncthreads();
  }

  const __bf16* __restrict__ ra = agg + (size_t)(t0 + m) * GF;
  const __bf16* __restrict__ rh = h   + (size_t)(t0 + m) * GF;

  v8f acc = {0.f, 0.f, 0.f, 0.f, 0.f, 0.f, 0.f, 0.f};

  #pragma unroll
  for (int kc = 0; kc < 8; ++kc) {
    const __bf16* src = (kc < 4) ? (ra + (kc << 5)) : (rh + ((kc - 4) << 5));
    v8bf lo = *reinterpret_cast<const v8bf*>(src + (g << 4));
    v8bf hi = *reinterpret_cast<const v8bf*>(src + 8 + (g << 4));
    v16bf a;
    #pragma unroll
    for (int e = 0; e < 8; ++e) { a[e] = lo[e]; a[8 + e] = hi[e]; }
    // B from LDS: row o0+m, swizzled 32 B pair (g + 2*kc) ^ m
    const __bf16* rwl = wsm + (size_t)(o0 + m) * 256 + (((unsigned)(g + (kc << 1)) ^ (unsigned)m) << 4);
    v16bf b = *reinterpret_cast<const v16bf*>(rwl);
    acc = __builtin_amdgcn_wmma_f32_16x16x32_bf16(false, a, false, b,
                                                  (short)0, acc, false, false);
  }

  float bb = bias[o0 + m];
  #pragma unroll
  for (int r = 0; r < 8; ++r) {
    int t = t0 + r + (g << 3);
    float v = acc[r] + bb;
    if (RELU) v = fmaxf(v, 0.0f);
    if (TO_BF16) {
      __bf16 q = (__bf16)v;
      hout[(size_t)t * GF + o0 + m] = q;          // row-major for next dense A
      hToutT[(size_t)(o0 + m) * GN + t] = q;      // feature-major for next SpMM B
    } else {
      fout[(size_t)t * FO + o0 + m] = v;          // final f32 output
    }
  }
}

// ---------------------------------------------------------------- launcher
extern "C" void kernel_launch(void* const* d_in, const int* in_sizes, int n_in,
                              void* d_out, int out_size, void* d_ws, size_t ws_size,
                              hipStream_t stream) {
  (void)in_sizes; (void)n_in; (void)out_size; (void)ws_size;

  const float* x   = (const float*)d_in[0];
  const float* adj = (const float*)d_in[1];
  const float* Wl0 = (const float*)d_in[2];
  const float* b0  = (const float*)d_in[3];
  const float* Wr0 = (const float*)d_in[4];
  const float* Wl1 = (const float*)d_in[5];
  const float* b1  = (const float*)d_in[6];
  const float* Wr1 = (const float*)d_in[7];
  const float* Wl2 = (const float*)d_in[8];
  const float* b2  = (const float*)d_in[9];
  const float* Wr2 = (const float*)d_in[10];
  float* out = (float*)d_out;

  char* p = (char*)d_ws;
  auto carve = [&](size_t bytes) -> char* {
    char* q = p; p += (bytes + 255) & ~size_t(255); return q;
  };
  unsigned* bits = (unsigned*)carve((size_t)GN * GNW * 4);   // 32 MB, L2-resident
  unsigned* deg  = (unsigned*)carve((size_t)GN * 4);
  float*    dinv = (float*)   carve((size_t)GN * 4);
  __bf16* h0   = (__bf16*)carve((size_t)GN * GF * 2);
  __bf16* h0T  = (__bf16*)carve((size_t)GN * GF * 2);
  __bf16* h1   = (__bf16*)carve((size_t)GN * GF * 2);
  __bf16* h1T  = (__bf16*)carve((size_t)GN * GF * 2);
  __bf16* h2   = (__bf16*)carve((size_t)GN * GF * 2);
  __bf16* h2T  = (__bf16*)carve((size_t)GN * GF * 2);
  __bf16* aggb = (__bf16*)carve((size_t)GN * GF * 2);
  __bf16* wc0  = (__bf16*)carve((size_t)128 * 256 * 2);
  __bf16* wc1  = (__bf16*)carve((size_t)128 * 256 * 2);
  __bf16* wc2  = (__bf16*)carve((size_t)64  * 256 * 2);

  // preprocessing: adj read exactly once
  k_zero_u32<<<GN / 256, 256, 0, stream>>>(deg, GN);
  dim3 gp(GN / 256, GN / 32);
  k_pack_mask<<<gp, 256, 0, stream>>>(adj, bits, deg);
  k_deg_inv<<<GN / 256, 256, 0, stream>>>(deg, dinv);
  k_cvt_x<<<(GN * GF) / 256, 256, 0, stream>>>(x, h0, h0T);
  k_build_wcat<<<(128 * 256) / 256, 256, 0, stream>>>(Wl0, Wr0, wc0, 128);
  k_build_wcat<<<(128 * 256) / 256, 256, 0, stream>>>(Wl1, Wr1, wc1, 128);
  k_build_wcat<<<(64  * 256) / 256, 256, 0, stream>>>(Wl2, Wr2, wc2, 64);

  // layer 0
  k_spmm<<<GN / 16, 256, 0, stream>>>(bits, h0T, dinv, aggb);
  k_dense<true,  true,  128><<<GN / 16, 256, 0, stream>>>(aggb, h0, wc0, b0, h1, h1T, nullptr);
  // layer 1
  k_spmm<<<GN / 16, 256, 0, stream>>>(bits, h1T, dinv, aggb);
  k_dense<true,  true,  128><<<GN / 16, 256, 0, stream>>>(aggb, h1, wc1, b1, h2, h2T, nullptr);
  // layer 2 (no relu, f32 out, Fo=64 -> 4 waves)
  k_spmm<<<GN / 16, 256, 0, stream>>>(bits, h2T, dinv, aggb);
  k_dense<false, false, 64 ><<<GN / 16, 128, 0, stream>>>(aggb, h2, wc2, b2, nullptr, nullptr, out);
}
